// GraphSAGE_58342835749491
// MI455X (gfx1250) — matrix operand order repaired and checked by reading, not compiled, not run on hardware
//
#include <hip/hip_runtime.h>
#include <hip/hip_bf16.h>

// ---------------------------------------------------------------------------
// GraphSAGE (3-layer, mean aggregator) for gfx1250 / MI455X.
//   layer: out = x @ W_self + (segsum(x[src]) / max(deg,1)) @ W_neigh + b
// Scatter-bound (~4-5 GB/pass; agg arrays are L2-resident at 192MB L2), so the
// dense parts run on v_wmma_f32_16x16x32_bf16 and stay off the critical path.
// ---------------------------------------------------------------------------

typedef __bf16 bf16_t;
typedef __attribute__((ext_vector_type(16))) __bf16 v16bf;
typedef __attribute__((ext_vector_type(8)))  float  v8f;

// -------------------------- degree kernels ---------------------------------
__global__ void sage_degree(const int* __restrict__ dst, float* __restrict__ deg, int E) {
    int e = blockIdx.x * blockDim.x + threadIdx.x;
    if (e < E) unsafeAtomicAdd(&deg[dst[e]], 1.0f);
}

__global__ void sage_invdeg(float* __restrict__ deg, int N) {
    int n = blockIdx.x * blockDim.x + threadIdx.x;
    if (n < N) deg[n] = 1.0f / fmaxf(deg[n], 1.0f);
}

// -------------------------- edge scatter -----------------------------------
// One thread per (edge, 4-channel chunk): float4 gather from x[src], four
// global_atomic_add_f32 into agg[dst]. Consecutive threads cover consecutive
// channels of one edge -> coalesced gathers; atomics land in L2.
template<int F>
__global__ void sage_scatter(const float* __restrict__ x,
                             const int*  __restrict__ src,
                             const int*  __restrict__ dst,
                             float*      __restrict__ agg,
                             int E)
{
    constexpr int CH = F / 4;
    long t = (long)blockIdx.x * blockDim.x + threadIdx.x;
    if (t >= (long)E * CH) return;
    int e = (int)(t / CH);
    int c = (int)(t % CH) * 4;
    int s = src[e];
    int d = dst[e];
    const float4 v = *(const float4*)(x + (size_t)s * F + c);
    float* p = agg + (size_t)d * F + c;
    unsafeAtomicAdd(p + 0, v.x);
    unsafeAtomicAdd(p + 1, v.y);
    unsafeAtomicAdd(p + 2, v.z);
    unsafeAtomicAdd(p + 3, v.w);
}

// -------------------------- fused SAGE GEMM --------------------------------
// Block = 16-row tile of nodes; wave w = 16-col tile of outputs.
// acc += A_self x W_self + A_neigh x W_neigh over K in steps of 32, then +bias
// and optional ReLU. A tiles are staged through LDS as bf16 (float4 global
// loads, packed 8B ds stores), shared by all column waves. All weight (B)
// fragments are preloaded into registers before the K loop; the K loop is
// fully unrolled so B stays constant-indexed.
//
// Fragment layouts used here (contiguous k) (wave32, 16x16x32 bf16):
//   A: lane = (g<<4)|m, element i -> A[m][i + 16*g]
//   B: lane = (g<<4)|n, element i -> B[i + 16*g][n]
//   C/D: vgpr r, lane = (g<<4)|n   -> C[r + 8*g][n]
template<int IN_F, int OUT_F, bool RELU>
__global__ __launch_bounds__((OUT_F / 16) * 32)
void sage_gemm(const float* __restrict__ x,
               const float* __restrict__ agg,
               const float* __restrict__ invdeg,
               const float* __restrict__ wself,
               const float* __restrict__ wneigh,
               const float* __restrict__ bias,
               float* __restrict__ out,
               int N)
{
    constexpr int NW = OUT_F / 16;   // waves per block
    constexpr int NT = NW * 32;      // threads per block
    constexpr int KT = IN_F / 32;    // K tiles

    __shared__ bf16_t ldsA[2][16][32];   // [0]=self, [1]=neigh (pre-scaled)

    const int lane = threadIdx.x & 31;
    const int wave = threadIdx.x >> 5;
    const int row0 = blockIdx.x * 16;
    const int g    = lane >> 4;      // half-wave group
    const int mn   = lane & 15;      // A-row / B-col within tile
    const int col0 = wave * 16;

    // ---- preload ALL weight fragments into registers (<= 64 VGPRs) ----
    v16bf bs[KT], bn[KT];
    #pragma unroll
    for (int kt = 0; kt < KT; ++kt) {
        const float* wc0 = wself  + (size_t)(kt * 32 + 16 * g) * OUT_F + col0 + mn;
        const float* wc1 = wneigh + (size_t)(kt * 32 + 16 * g) * OUT_F + col0 + mn;
        #pragma unroll
        for (int i = 0; i < 16; ++i) {
            bs[kt][i] = (bf16_t)wc0[(size_t)i * OUT_F];
            bn[kt][i] = (bf16_t)wc1[(size_t)i * OUT_F];
        }
    }

    v8f acc = {};

    #pragma unroll
    for (int kt = 0; kt < KT; ++kt) {
        const int kbase = kt * 32;

        // ---- cooperative LDS stage: 256 float4 chunks (2 matrices x 128) ----
        // chunk q of matrix mtx covers row r = q/8, cols [4*(q%8), +4)
        #pragma unroll
        for (int idx = threadIdx.x; idx < 256; idx += NT) {
            const int mtx = idx >> 7;           // 0 = self(x), 1 = neigh(agg)
            const int q   = idx & 127;
            const int r   = q >> 3;
            const int c   = (q & 7) << 2;
            int row = row0 + r;
            if (row >= N) row = N - 1;          // clamp (keeps EXEC full)
            const float* sp = (mtx ? agg : x) + (size_t)row * IN_F + kbase + c;
            float4 v = *(const float4*)sp;
            if (mtx) {
                const float s = invdeg[row];
                v.x *= s; v.y *= s; v.z *= s; v.w *= s;
            }
            bf16_t t4[4] = {(bf16_t)v.x, (bf16_t)v.y, (bf16_t)v.z, (bf16_t)v.w};
            *(uint2*)&ldsA[mtx][r][c] = *(const uint2*)t4;   // 8B ds store
        }
        __syncthreads();

        // ---- A fragments from LDS ----
        v16bf a0, a1;
        #pragma unroll
        for (int i = 0; i < 16; ++i) {
            const int k = i + 16 * g;
            a0[i] = ldsA[0][mn][k];
            a1[i] = ldsA[1][mn][k];
        }

        // ---- two chained WMMAs (D->C accumulation, no A/B hazard) ----
        acc = __builtin_amdgcn_wmma_f32_16x16x32_bf16(false, a0, false, bs[kt],
                                                      (short)0, acc, false, false);
        acc = __builtin_amdgcn_wmma_f32_16x16x32_bf16(false, a1, false, bn[kt],
                                                      (short)0, acc, false, false);
        __syncthreads();
    }

    const float bv = bias[col0 + mn];
    #pragma unroll
    for (int r = 0; r < 8; ++r) {
        const int row = row0 + r + 8 * g;
        float v = acc[r] + bv;
        if (RELU) v = fmaxf(v, 0.0f);
        if (row < N) out[(size_t)row * OUT_F + col0 + mn] = v;
    }
}

// -------------------------- row softmax (64 wide) --------------------------
__global__ void sage_softmax64(float* __restrict__ out, int N) {
    int wid  = (int)(((long)blockIdx.x * blockDim.x + threadIdx.x) >> 5);
    int lane = threadIdx.x & 31;
    if (wid >= N) return;
    float* row = out + (size_t)wid * 64;
    float v0 = row[lane], v1 = row[lane + 32];
    float m = fmaxf(v0, v1);
    #pragma unroll
    for (int off = 16; off; off >>= 1) m = fmaxf(m, __shfl_xor(m, off, 32));
    float e0 = __expf(v0 - m), e1 = __expf(v1 - m);
    float s = e0 + e1;
    #pragma unroll
    for (int off = 16; off; off >>= 1) s += __shfl_xor(s, off, 32);
    float inv = 1.0f / s;
    row[lane]      = e0 * inv;
    row[lane + 32] = e1 * inv;
}

// ---------------------------------------------------------------------------
extern "C" void kernel_launch(void* const* d_in, const int* in_sizes, int n_in,
                              void* d_out, int out_size, void* d_ws, size_t ws_size,
                              hipStream_t stream) {
    const float* x    = (const float*)d_in[0];
    const int*   src  = (const int*)  d_in[1];
    const int*   dst  = (const int*)  d_in[2];
    const float* w1s  = (const float*)d_in[3];
    const float* w1n  = (const float*)d_in[4];
    const float* b1   = (const float*)d_in[5];
    const float* w2s  = (const float*)d_in[6];
    const float* w2n  = (const float*)d_in[7];
    const float* b2   = (const float*)d_in[8];
    const float* w3s  = (const float*)d_in[9];
    const float* w3n  = (const float*)d_in[10];
    const float* b3   = (const float*)d_in[11];
    float* out = (float*)d_out;

    const int N = in_sizes[0] / 128;   // 100000
    const int E = in_sizes[1];         // 1600000

    // workspace layout (floats): invdeg | agg(N*128) | h1(N*128) | h2(N*64)
    float* ws  = (float*)d_ws;
    float* deg = ws;
    float* agg = ws + (((size_t)N + 255) & ~(size_t)255);
    float* h1  = agg + (size_t)N * 128;
    float* h2  = h1  + (size_t)N * 128;

    const int rowTiles = (N + 15) / 16;

    // degrees (dst fixed -> compute once, reuse for all 3 layers)
    hipMemsetAsync(deg, 0, sizeof(float) * (size_t)N, stream);
    sage_degree<<<(E + 255) / 256, 256, 0, stream>>>(dst, deg, E);
    sage_invdeg<<<(N + 255) / 256, 256, 0, stream>>>(deg, N);

    // ---- layer 1: 128 -> 128, ReLU ----
    hipMemsetAsync(agg, 0, sizeof(float) * (size_t)N * 128, stream);
    {
        long total = (long)E * 32;
        sage_scatter<128><<<(int)((total + 255) / 256), 256, 0, stream>>>(x, src, dst, agg, E);
    }
    sage_gemm<128, 128, true><<<rowTiles, 256, 0, stream>>>(x, agg, deg, w1s, w1n, b1, h1, N);

    // ---- layer 2: 128 -> 64, ReLU ----
    hipMemsetAsync(agg, 0, sizeof(float) * (size_t)N * 128, stream);
    {
        long total = (long)E * 32;
        sage_scatter<128><<<(int)((total + 255) / 256), 256, 0, stream>>>(h1, src, dst, agg, E);
    }
    sage_gemm<128, 64, true><<<rowTiles, 128, 0, stream>>>(h1, agg, deg, w2s, w2n, b2, h2, N);

    // ---- layer 3: 64 -> 64, logits ----
    hipMemsetAsync(agg, 0, sizeof(float) * (size_t)N * 64, stream);
    {
        long total = (long)E * 16;
        sage_scatter<64><<<(int)((total + 255) / 256), 256, 0, stream>>>(h2, src, dst, agg, E);
    }
    sage_gemm<64, 64, false><<<rowTiles, 128, 0, stream>>>(h2, agg, deg, w3s, w3n, b3, out, N);

    // ---- softmax over 64 classes, one wave per node ----
    {
        long threads = (long)N * 32;
        sage_softmax64<<<(int)((threads + 255) / 256), 256, 0, stream>>>(out, N);
    }
}
